// Bed_Poture_Heat_21_38362647888451
// MI455X (gfx1250) — matrix operand order repaired and checked by reading, not compiled, not run on hardware
//
#include <hip/hip_runtime.h>

typedef __attribute__((ext_vector_type(2)))  float  v2f;
typedef __attribute__((ext_vector_type(8)))  float  v8f;
typedef __attribute__((ext_vector_type(16))) __bf16 v16bf;

#define CH_LEN 2048
#define N_WIN 16
#define WIN 256
#define STRIDE 119
#define BATCH_TILE 16

#define H_PITCH 516   // 516 % 64 == 4 -> conflict-free LDS fragment reads
#define G_PITCH 260   // 260 % 64 == 4

#define W1_ELEMS (3 * N_WIN * WIN * 64)      // 786432
#define W1_CHUNKS (3 * N_WIN * 8 * 4 * 32)   // 49152 chunks of 16 bf16
#define W2P_ELEMS (4 * 64 * 2 * 64)          // 32768 float2
#define W3P_ELEMS (64 * 2 * 256)             // 32768 float2 (N padded to 256)

__device__ __forceinline__ v8f wmma4_f32(v2f a, v2f b, v8f c) {
  return __builtin_amdgcn_wmma_f32_16x16x4_f32(false, a, false, b,
                                               (short)0, c, false, false);
}
__device__ __forceinline__ v8f wmma32_bf16(v16bf a, v16bf b, v8f c) {
  return __builtin_amdgcn_wmma_f32_16x16x32_bf16(false, a, false, b,
                                                 (short)0, c, false, false);
}

// ===========================================================================
// Weight repack: W1 -> bf16 hi/lo in per-lane WMMA fragment order,
//                W2/W3 -> float2 k-pairs matching the f32 WMMA B layout.
// ===========================================================================
__global__ void pack_weights_kernel(const float* __restrict__ W1,
                                    const float* __restrict__ W2,
                                    const float* __restrict__ W3,
                                    __bf16* __restrict__ W1hi,
                                    __bf16* __restrict__ W1lo,
                                    v2f* __restrict__ W2p,
                                    v2f* __restrict__ W3p) {
  const int tid = blockIdx.x * blockDim.x + threadIdx.x;
  const int nth = gridDim.x * blockDim.x;

  // ---- W1: chunk p = ((((c*16+w)*8+kk)*4+nt)*32+lane), 16 bf16 per chunk --
  for (int p = tid; p < W1_CHUNKS; p += nth) {
    const int lane = p & 31;
    const int nt   = (p >> 5) & 3;
    const int kk   = (p >> 7) & 7;
    const int w    = (p >> 10) & 15;
    const int c    = p >> 14;
    const int h    = lane >> 4;
    const int n    = nt * 16 + (lane & 15);
    const float* src = W1 + (((size_t)c * N_WIN + w) * WIN) * 64 + n;
    __bf16* dhi = W1hi + (size_t)p * 16;
    __bf16* dlo = W1lo + (size_t)p * 16;
    #pragma unroll
    for (int j = 0; j < 16; ++j) {
      const int K = kk * 32 + (j < 8 ? h * 8 + j : 16 + h * 8 + (j - 8));  // two 8-K runs per lane (table order)
      const float f = src[(size_t)K * 64];
      const __bf16 hi = (__bf16)f;
      dhi[j] = hi;
      dlo[j] = (__bf16)(f - (float)hi);
    }
  }
  // ---- W2p[((g*64+kk)*2+kh)*64 + n] = {W2[g][4kk+2kh][n], [..+1][n]} ------
  for (int p = tid; p < W2P_ELEMS; p += nth) {
    const int n  = p & 63;
    const int kh = (p >> 6) & 1;
    const int kk = (p >> 7) & 63;
    const int g  = p >> 13;
    const int k  = kk * 4 + kh * 2;
    v2f v;
    v.x = W2[((size_t)g * WIN + k) * 64 + n];
    v.y = W2[((size_t)g * WIN + k + 1) * 64 + n];
    W2p[p] = v;
  }
  // ---- W3p[(kk*2+kh)*256 + n], zero-padded at n==255 ----------------------
  for (int p = tid; p < W3P_ELEMS; p += nth) {
    const int n  = p & 255;
    const int kh = (p >> 8) & 1;
    const int kk = p >> 9;
    const int k  = kk * 4 + kh * 2;
    v2f v;
    v.x = (n < 255) ? W3[(size_t)k * 255 + n] : 0.0f;
    v.y = (n < 255) ? W3[(size_t)(k + 1) * 255 + n] : 0.0f;
    W3p[p] = v;
  }
}

// ===========================================================================
// Main fused kernel: layer1 = bf16x3 split-precision WMMA (16x16x32) with
// one window per wave (A fragment converted once, reused by 4 n-tiles),
// layers 2/3 = exact f32 WMMA (16x16x4). One 16-row batch tile per WG.
// ===========================================================================
__global__ __launch_bounds__(256) void fused_bf16x3_kernel(
    const float* __restrict__ x,       // [B][3*2048]
    const float* __restrict__ b1,      // [3][16][64]
    const float* __restrict__ b2,      // [4][64]
    const float* __restrict__ b3,      // [255]
    const __bf16* __restrict__ W1hi,   // packed fragments
    const __bf16* __restrict__ W1lo,
    const v2f* __restrict__ W2p,
    const v2f* __restrict__ W3p,
    float* __restrict__ out)           // [B][255]
{
  __shared__ float sh_h[BATCH_TILE * H_PITCH];
  __shared__ float sh_g[BATCH_TILE * G_PITCH];

  const int b0   = blockIdx.x * BATCH_TILE;
  const int tid  = threadIdx.x;
  const int lane = tid & 31;
  const int wv   = tid >> 5;
  const int ml   = lane & 15;
  const int kh   = lane >> 4;

  for (int gp = 0; gp < 2; ++gp) {
    const int wbase = gp * 8;

    // ---- Layer 1 (bf16x3): wave wv owns window wbase+wv, all 4 n-tiles ----
    {
      const int w = wbase + wv;
      v8f acc[4] = {{}, {}, {}, {}};

      for (int c = 0; c < 3; ++c) {
        const float* Arow = x + (size_t)(b0 + ml) * (3 * CH_LEN)
                              + c * CH_LEN + STRIDE * w;
        // per-(c,w) base chunk index (nt added below)
        const size_t cw = (((size_t)c * N_WIN + w) * 8);
        for (int kk = 0; kk < 8; ++kk) {
          // A fragment: two contiguous 8-float runs, converted ONCE per kk
          const float* Ap = Arow + kk * 32 + kh * 8;
          float xs[16];
          #pragma unroll
          for (int j = 0; j < 8; ++j) xs[j] = Ap[j];
          #pragma unroll
          for (int j = 0; j < 8; ++j) xs[8 + j] = Ap[16 + j];
          v16bf ahi, alo;
          #pragma unroll
          for (int j = 0; j < 16; ++j) {
            const float f = xs[j];
            const __bf16 hi = (__bf16)f;
            ahi[j] = hi;
            alo[j] = (__bf16)(f - (float)hi);
          }
          // 4 n-tiles share this A fragment
          const size_t ck = ((cw + kk) * 4) * 32 + lane;
          #pragma unroll
          for (int nt = 0; nt < 4; ++nt) {
            const size_t chunk = ck + (size_t)nt * 32;
            const v16bf bhi = *(const v16bf*)(W1hi + chunk * 16);
            const v16bf blo = *(const v16bf*)(W1lo + chunk * 16);
            acc[nt] = wmma32_bf16(alo, bhi, acc[nt]);   // lo*hi
            acc[nt] = wmma32_bf16(ahi, blo, acc[nt]);   // hi*lo
            acc[nt] = wmma32_bf16(ahi, bhi, acc[nt]);   // hi*hi
          }
        }
      }
      // epilogue: mean over channels + (mean of b1), store 4 tiles
      #pragma unroll
      for (int nt = 0; nt < 4; ++nt) {
        const int n = nt * 16 + ml;
        const float bias = (b1[(0 * N_WIN + w) * 64 + n] +
                            b1[(1 * N_WIN + w) * 64 + n] +
                            b1[(2 * N_WIN + w) * 64 + n]) * (1.0f / 3.0f);
        #pragma unroll
        for (int v = 0; v < 8; ++v) {
          const int row = v + 8 * kh;
          sh_h[row * H_PITCH + wv * 64 + nt * 16 + ml] =
              acc[nt][v] * (1.0f / 3.0f) + bias;
        }
      }
    }
    __syncthreads();

    // ---- Layer 2 (f32 WMMA): groups 2*gp, 2*gp+1; one n-tile per wave -----
    {
      const int gl = wv >> 2;
      const int g  = gp * 2 + gl;
      const int nt = wv & 3;
      const int n  = nt * 16 + ml;

      v8f acc = {};
      const float* Ab = sh_h + ml * H_PITCH + gl * 256;
      #pragma unroll 8
      for (int kk = 0; kk < WIN / 4; ++kk) {
        const int k = kk * 4 + kh * 2;
        v2f a = *(const v2f*)(Ab + k);
        v2f b = W2p[(((size_t)g * 64 + kk) * 2 + kh) * 64 + n];
        acc = wmma4_f32(a, b, acc);
      }
      const float bias = b2[g * 64 + n];
      #pragma unroll
      for (int v = 0; v < 8; ++v) {
        const int row = v + 8 * kh;
        sh_g[row * G_PITCH + g * 64 + n] = acc[v] + bias;
      }
    }
    __syncthreads();
  }

  // ---- Layer 3 (f32 WMMA): N padded to 256; 2 n-tiles per wave ------------
  for (int nt = wv; nt < 16; nt += 8) {
    const int n = nt * 16 + ml;
    v8f acc = {};
    const float* Ab = sh_g + ml * G_PITCH;
    #pragma unroll 8
    for (int kk = 0; kk < WIN / 4; ++kk) {
      const int k = kk * 4 + kh * 2;
      v2f a = *(const v2f*)(Ab + k);
      v2f b = W3p[((size_t)kk * 2 + kh) * 256 + n];
      acc = wmma4_f32(a, b, acc);
    }
    if (n < 255) {
      const float bias = b3[n];
      #pragma unroll
      for (int v = 0; v < 8; ++v) {
        const int row = v + 8 * kh;
        out[(size_t)(b0 + row) * 255 + n] = acc[v] + bias;
      }
    }
  }
}

// ===========================================================================
// Fallback: all-f32 WMMA path (no workspace needed).
// ===========================================================================
__global__ __launch_bounds__(256) void fused_f32_kernel(
    const float* __restrict__ x, const float* __restrict__ W1,
    const float* __restrict__ b1, const float* __restrict__ W2,
    const float* __restrict__ b2, const float* __restrict__ W3,
    const float* __restrict__ b3, float* __restrict__ out)
{
  __shared__ float sh_h[BATCH_TILE * H_PITCH];
  __shared__ float sh_g[BATCH_TILE * G_PITCH];

  const int b0   = blockIdx.x * BATCH_TILE;
  const int tid  = threadIdx.x;
  const int lane = tid & 31;
  const int wv   = tid >> 5;
  const int ml   = lane & 15;
  const int kh   = lane >> 4;

  for (int gp = 0; gp < 2; ++gp) {
    const int wbase = gp * 8;
    for (int t = wv; t < 32; t += 8) {
      const int wl = t >> 2, nt = t & 3;
      const int w = wbase + wl;
      const int n = nt * 16 + ml;
      v8f acc = {};
      for (int c = 0; c < 3; ++c) {
        const float* Arow  = x + (size_t)(b0 + ml) * (3 * CH_LEN)
                               + c * CH_LEN + STRIDE * w;
        const float* Bbase = W1 + (((size_t)c * N_WIN + w) * WIN) * 64 + n;
        #pragma unroll 8
        for (int kk = 0; kk < WIN / 4; ++kk) {
          const int k = kk * 4 + kh * 2;
          v2f a, b;
          a.x = Arow[k];  a.y = Arow[k + 1];
          b.x = Bbase[(size_t)k * 64];  b.y = Bbase[(size_t)(k + 1) * 64];
          acc = wmma4_f32(a, b, acc);
        }
      }
      const float bias = (b1[(0 * N_WIN + w) * 64 + n] +
                          b1[(1 * N_WIN + w) * 64 + n] +
                          b1[(2 * N_WIN + w) * 64 + n]) * (1.0f / 3.0f);
      #pragma unroll
      for (int v = 0; v < 8; ++v)
        sh_h[(v + 8 * kh) * H_PITCH + wl * 64 + nt * 16 + ml] =
            acc[v] * (1.0f / 3.0f) + bias;
    }
    __syncthreads();
    {
      const int gl = wv >> 2;
      const int g  = gp * 2 + gl;
      const int nt = wv & 3;
      const int n  = nt * 16 + ml;
      v8f acc = {};
      const float* Bbase = W2 + ((size_t)g * WIN) * 64 + n;
      const float* Ab    = sh_h + ml * H_PITCH + gl * 256;
      #pragma unroll 8
      for (int kk = 0; kk < WIN / 4; ++kk) {
        const int k = kk * 4 + kh * 2;
        v2f a = *(const v2f*)(Ab + k);
        v2f b;
        b.x = Bbase[(size_t)k * 64];  b.y = Bbase[(size_t)(k + 1) * 64];
        acc = wmma4_f32(a, b, acc);
      }
      const float bias = b2[g * 64 + n];
      #pragma unroll
      for (int v = 0; v < 8; ++v)
        sh_g[(v + 8 * kh) * G_PITCH + g * 64 + n] = acc[v] + bias;
    }
    __syncthreads();
  }
  for (int nt = wv; nt < 16; nt += 8) {
    const int n = nt * 16 + ml;
    const bool valid = (n < 255);
    v8f acc = {};
    const float* Ab = sh_g + ml * G_PITCH;
    #pragma unroll 8
    for (int kk = 0; kk < WIN / 4; ++kk) {
      const int k = kk * 4 + kh * 2;
      v2f a = *(const v2f*)(Ab + k);
      v2f b;
      b.x = valid ? W3[(size_t)k * 255 + n] : 0.0f;
      b.y = valid ? W3[(size_t)(k + 1) * 255 + n] : 0.0f;
      acc = wmma4_f32(a, b, acc);
    }
    if (valid) {
      const float bias = b3[n];
      #pragma unroll
      for (int v = 0; v < 8; ++v)
        out[(size_t)(b0 + v + 8 * kh) * 255 + n] = acc[v] + bias;
    }
  }
}

extern "C" void kernel_launch(void* const* d_in, const int* in_sizes, int n_in,
                              void* d_out, int out_size, void* d_ws, size_t ws_size,
                              hipStream_t stream) {
  const float* x  = (const float*)d_in[0];
  const float* W1 = (const float*)d_in[1];
  const float* b1 = (const float*)d_in[2];
  const float* W2 = (const float*)d_in[3];
  const float* b2 = (const float*)d_in[4];
  const float* W3 = (const float*)d_in[5];
  const float* b3 = (const float*)d_in[6];
  float* out = (float*)d_out;

  const int B = in_sizes[0] / (3 * CH_LEN);   // 8192
  dim3 grid(B / BATCH_TILE);                  // 512 workgroups
  dim3 block(256);                            // 8 waves (wave32)

  const size_t need = (size_t)2 * W1_ELEMS * sizeof(__bf16)   // W1 hi+lo
                    + (size_t)W2P_ELEMS * sizeof(v2f)
                    + (size_t)W3P_ELEMS * sizeof(v2f);        // ~3.7 MB

  if (ws_size >= need) {
    __bf16* W1hi = (__bf16*)d_ws;
    __bf16* W1lo = W1hi + W1_ELEMS;
    v2f*    W2p  = (v2f*)(W1lo + W1_ELEMS);
    v2f*    W3p  = W2p + W2P_ELEMS;

    hipLaunchKernelGGL(pack_weights_kernel, dim3(256), dim3(256), 0, stream,
                       W1, W2, W3, W1hi, W1lo, W2p, W3p);
    hipLaunchKernelGGL(fused_bf16x3_kernel, grid, block, 0, stream,
                       x, b1, b2, b3, W1hi, W1lo, W2p, W3p, out);
  } else {
    hipLaunchKernelGGL(fused_f32_kernel, grid, block, 0, stream,
                       x, W1, b1, W2, b2, W3, b3, out);
  }
}
